// CardRecommendationSAGE_37160057045127
// MI455X (gfx1250) — matrix operand order repaired and checked by reading, not compiled, not run on hardware
//
#include <hip/hip_runtime.h>

// ---------------------------------------------------------------------------
// CardRecommendationSAGE for MI455X (gfx1250, wave32, WMMA bf16)
// ---------------------------------------------------------------------------

typedef __attribute__((ext_vector_type(16))) __bf16 v16bf;
typedef __attribute__((ext_vector_type(8)))  __bf16 v8bf;
typedef __attribute__((ext_vector_type(8)))  float  v8f;

#define NNODES 100000
#define NEDGES 1600000
#define FIN    128
#define DIN    129        // FIN + input_cards indicator
#define HDIM   128
#define KPAD   160        // pad K to 5 x 32 for clean WMMA steps
#define NCARDS 110
#define CPAD   112        // NCARDS padded to 7 x 16 column tiles

// ----------------------------- utility kernels -----------------------------

__global__ void zerof_k(float* __restrict__ p, int n) {
    int i = blockIdx.x * blockDim.x + threadIdx.x;
    if (i < n) p[i] = 0.0f;
}

// Convert f32 weight [rows, cols] -> bf16 [rowsPad(grid), KPAD] zero-padded.
__global__ void convert_pad_k(const float* __restrict__ W, __bf16* __restrict__ o,
                              int rows, int cols, int total) {
    int i = blockIdx.x * blockDim.x + threadIdx.x;
    if (i >= total) return;
    int r = i / KPAD;
    int c = i - r * KPAD;
    float v = (r < rows && c < cols) ? W[r * cols + c] : 0.0f;
    o[i] = (__bf16)v;
}

// h0[n, :] = [ x[n, 0:128] | (float)cards[n] | zeros ]  (bf16, stride KPAD)
__global__ void build_h0_k(const float* __restrict__ x, const int* __restrict__ cards,
                           __bf16* __restrict__ h0) {
    int i = blockIdx.x * blockDim.x + threadIdx.x;
    if (i >= NNODES * KPAD) return;
    int n = i / KPAD;
    int c = i - n * KPAD;
    float v = (c < FIN) ? x[n * FIN + c] : ((c == FIN) ? (float)cards[n] : 0.0f);
    h0[i] = (__bf16)v;
}

__global__ void degree_k(const long long* __restrict__ ei, float* __restrict__ deg) {
    int e = blockIdx.x * blockDim.x + threadIdx.x;
    if (e < NEDGES) atomicAdd(&deg[(size_t)ei[NEDGES + e]], 1.0f);
}

__global__ void deg_inv_k(const float* __restrict__ deg, float* __restrict__ dinv) {
    int i = blockIdx.x * blockDim.x + threadIdx.x;
    if (i < NNODES) dinv[i] = (deg[i] > 0.0f) ? (1.0f / deg[i]) : 0.0f;
}

// One block per edge; lane t scatter-adds feature t of h[src] into agg[dst].
__global__ void aggregate_k(const __bf16* __restrict__ Hsrc,
                            const long long* __restrict__ ei,
                            float* __restrict__ agg, int ncols) {
    const int e = blockIdx.x;
    const int t = threadIdx.x;
    if (t >= ncols) return;
    const size_t s = (size_t)ei[e];
    const size_t d = (size_t)ei[NEDGES + e];
    atomicAdd(&agg[d * KPAD + t], (float)Hsrc[s * KPAD + t]);
}

// meanA = bf16(agg * deg_inv)   (padding cols stay exactly zero)
__global__ void finalize_mean_k(const float* __restrict__ agg,
                                const float* __restrict__ dinv,
                                __bf16* __restrict__ meanA) {
    int i = blockIdx.x * blockDim.x + threadIdx.x;
    if (i >= NNODES * KPAD) return;
    meanA[i] = (__bf16)(agg[i] * dinv[i / KPAD]);
}

// ------------------------------ WMMA helpers --------------------------------
// A (16xK, bf16) per-lane fragment for K-step kb: lane holds row m = lane&15;
// lanes 0-15 take K = kb+0..15 (contiguous), lanes 16-31 take K = kb+16..31 (contiguous).
// B (Kx16 = W^T) fragment: lane holds column j (= W row j); lanes 0-15 take
// K = kb+0..15 (contiguous), lanes 16-31 take K = kb+16..31 (contiguous).

__device__ __forceinline__ v16bf load_a_frag(const __bf16* rowPtr, int kb, int aoff) {
    const __bf16* p = rowPtr + kb + aoff;
    v8bf lo = *(const v8bf*)(p);
    v8bf hi = *(const v8bf*)(p + 8);
    return __builtin_shufflevector(lo, hi, 0,1,2,3,4,5,6,7,8,9,10,11,12,13,14,15);
}

// ------------------------ fused dual-GEMM (SAGE layer) ----------------------
// out[m, j] = relu( sum_k mean[m,k]*Wl[j,k] + self[m,k]*Wr[j,k] + bl[j] + br[j] )
// Block = 256 threads = 8 waves; wave w computes node-tile blockIdx.x, h-tile w.

__global__ __launch_bounds__(256) void sage_gemm_dual_k(
        const __bf16* __restrict__ Aself, const __bf16* __restrict__ Amean,
        const __bf16* __restrict__ Wr,    const __bf16* __restrict__ Wl,
        const float*  __restrict__ br,    const float*  __restrict__ bl,
        __bf16* __restrict__ Hout, int kSteps) {
    const int lane = threadIdx.x & 31;
    const int wv   = threadIdx.x >> 5;       // h tile 0..7
    const int l15  = lane & 15;
    const int hi   = lane >> 4;
    const size_t m = (size_t)blockIdx.x * 16 + l15;   // A row this lane feeds
    const int    j = wv * 16 + l15;                   // output feature / W row
    const int aoff = hi * 16;
    const int boff = hi * 16;

    const __bf16* amRow = Amean + m * KPAD;
    const __bf16* asRow = Aself + m * KPAD;
    const __bf16* wlRow = Wl + (size_t)j * KPAD;
    const __bf16* wrRow = Wr + (size_t)j * KPAD;

    v8f c = {};
    for (int ks = 0; ks < kSteps; ++ks) {
        const int kb = ks * 32;
        v16bf am = load_a_frag(amRow, kb, aoff);
        v16bf as = load_a_frag(asRow, kb, aoff);
        v16bf bL = *(const v16bf*)(wlRow + kb + boff);
        v16bf bR = *(const v16bf*)(wrRow + kb + boff);
        c = __builtin_amdgcn_wmma_f32_16x16x32_bf16(false, am, false, bL,
                                                    (short)0, c, false, false);
        c = __builtin_amdgcn_wmma_f32_16x16x32_bf16(false, as, false, bR,
                                                    (short)0, c, false, false);
    }

    const float bias = bl[j] + br[j];
    const size_t rowBase = (size_t)blockIdx.x * 16 + (size_t)hi * 8;  // M = r + 8*hi
#pragma unroll
    for (int r = 0; r < 8; ++r) {
        float v = c[r] + bias;
        v = v > 0.0f ? v : 0.0f;                     // ReLU
        Hout[(rowBase + r) * KPAD + j] = (__bf16)v;
    }
}

// ------------------------------ output head --------------------------------
// logits[m, j] = sum_k h2[m,k]*Wo[j,k] + bo[j], j < 110, f32 out (stride 110)
// Block = 224 threads = 7 waves covering the 7 padded column tiles.

__global__ __launch_bounds__(224) void sage_gemm_out_k(
        const __bf16* __restrict__ A, const __bf16* __restrict__ W,
        const float* __restrict__ bias, float* __restrict__ Out, int kSteps) {
    const int lane = threadIdx.x & 31;
    const int wv   = threadIdx.x >> 5;       // col tile 0..6
    const int l15  = lane & 15;
    const int hi   = lane >> 4;
    const size_t m = (size_t)blockIdx.x * 16 + l15;
    const int    j = wv * 16 + l15;
    const int aoff = hi * 16;
    const int boff = hi * 16;

    const __bf16* aRow = A + m * KPAD;
    const __bf16* wRow = W + (size_t)j * KPAD;

    v8f c = {};
    for (int ks = 0; ks < kSteps; ++ks) {
        const int kb = ks * 32;
        v16bf av = load_a_frag(aRow, kb, aoff);
        v16bf bv = *(const v16bf*)(wRow + kb + boff);
        c = __builtin_amdgcn_wmma_f32_16x16x32_bf16(false, av, false, bv,
                                                    (short)0, c, false, false);
    }

    const float b = (j < NCARDS) ? bias[j] : 0.0f;
    const size_t rowBase = (size_t)blockIdx.x * 16 + (size_t)hi * 8;
    if (j < NCARDS) {
#pragma unroll
        for (int r = 0; r < 8; ++r)
            Out[(rowBase + r) * NCARDS + j] = c[r] + b;
    }
}

// ------------------------------- launcher ----------------------------------

extern "C" void kernel_launch(void* const* d_in, const int* in_sizes, int n_in,
                              void* d_out, int out_size, void* d_ws, size_t ws_size,
                              hipStream_t stream) {
    (void)in_sizes; (void)n_in; (void)out_size; (void)ws_size;

    const float*     x    = (const float*)d_in[0];
    const float*     Wl1  = (const float*)d_in[1];
    const float*     bl1  = (const float*)d_in[2];
    const float*     Wr1  = (const float*)d_in[3];
    const float*     br1  = (const float*)d_in[4];
    const float*     Wl2  = (const float*)d_in[5];
    const float*     bl2  = (const float*)d_in[6];
    const float*     Wr2  = (const float*)d_in[7];
    const float*     br2  = (const float*)d_in[8];
    const float*     Wo   = (const float*)d_in[9];
    const float*     bo   = (const float*)d_in[10];
    const long long* ei   = (const long long*)d_in[11];
    const int*       cards= (const int*)d_in[12];
    float*           out  = (float*)d_out;

    // workspace carve-up (256B aligned)
    char*  base = (char*)d_ws;
    size_t off  = 0;
    auto alloc = [&](size_t bytes) -> char* {
        off = (off + 255) & ~(size_t)255;
        char* p = base + off;
        off += bytes;
        return p;
    };
    float*  deg   = (float*) alloc((size_t)NNODES * 4);
    float*  dinv  = (float*) alloc((size_t)NNODES * 4);
    __bf16* h0    = (__bf16*)alloc((size_t)NNODES * KPAD * 2);  // reused as h2
    __bf16* h1    = (__bf16*)alloc((size_t)NNODES * KPAD * 2);
    __bf16* meanA = (__bf16*)alloc((size_t)NNODES * KPAD * 2);
    float*  agg   = (float*) alloc((size_t)NNODES * KPAD * 4);
    __bf16* wl1b  = (__bf16*)alloc((size_t)HDIM * KPAD * 2);
    __bf16* wr1b  = (__bf16*)alloc((size_t)HDIM * KPAD * 2);
    __bf16* wl2b  = (__bf16*)alloc((size_t)HDIM * KPAD * 2);
    __bf16* wr2b  = (__bf16*)alloc((size_t)HDIM * KPAD * 2);
    __bf16* wob   = (__bf16*)alloc((size_t)CPAD * KPAD * 2);

    const int T = 256;
    const int nodeTiles = NNODES / 16;        // 6250 exactly
    const int nPad = NNODES * KPAD;           // 16,000,000

    // init
    zerof_k<<<(NNODES + T - 1) / T, T, 0, stream>>>(deg, NNODES);
    zerof_k<<<(nPad + T - 1) / T, T, 0, stream>>>(agg, nPad);

    // weights -> bf16, K-padded
    convert_pad_k<<<(HDIM * KPAD + T - 1) / T, T, 0, stream>>>(Wl1, wl1b, HDIM, DIN,  HDIM * KPAD);
    convert_pad_k<<<(HDIM * KPAD + T - 1) / T, T, 0, stream>>>(Wr1, wr1b, HDIM, DIN,  HDIM * KPAD);
    convert_pad_k<<<(HDIM * KPAD + T - 1) / T, T, 0, stream>>>(Wl2, wl2b, HDIM, HDIM, HDIM * KPAD);
    convert_pad_k<<<(HDIM * KPAD + T - 1) / T, T, 0, stream>>>(Wr2, wr2b, HDIM, HDIM, HDIM * KPAD);
    convert_pad_k<<<(CPAD * KPAD + T - 1) / T, T, 0, stream>>>(Wo,  wob, NCARDS, HDIM, CPAD * KPAD);

    // h0 = [x | cards | 0]
    build_h0_k<<<(nPad + T - 1) / T, T, 0, stream>>>(x, cards, h0);

    // degrees
    degree_k<<<(NEDGES + T - 1) / T, T, 0, stream>>>(ei, deg);
    deg_inv_k<<<(NNODES + T - 1) / T, T, 0, stream>>>(deg, dinv);

    // ---- layer 1 ----
    aggregate_k<<<NEDGES, KPAD, 0, stream>>>(h0, ei, agg, DIN);
    finalize_mean_k<<<(nPad + T - 1) / T, T, 0, stream>>>(agg, dinv, meanA);
    sage_gemm_dual_k<<<nodeTiles, 256, 0, stream>>>(h0, meanA, wr1b, wl1b, br1, bl1,
                                                    h1, KPAD / 32 /* 5 */);

    // ---- layer 2 ----
    zerof_k<<<(nPad + T - 1) / T, T, 0, stream>>>(agg, nPad);
    aggregate_k<<<NEDGES, KPAD, 0, stream>>>(h1, ei, agg, HDIM);
    finalize_mean_k<<<(nPad + T - 1) / T, T, 0, stream>>>(agg, dinv, meanA);
    sage_gemm_dual_k<<<nodeTiles, 256, 0, stream>>>(h1, meanA, wr2b, wl2b, br2, bl2,
                                                    h0 /* h2 */, HDIM / 32 /* 4 */);

    // ---- output head ----
    sage_gemm_out_k<<<nodeTiles, 224, 0, stream>>>(h0, wob, bo, out, HDIM / 32);
}
